// C4MoEVM_62380105007560
// MI455X (gfx1250) — matrix operand order repaired and checked by reading, not compiled, not run on hardware
//
#include <hip/hip_runtime.h>
#include <hip/hip_bf16.h>

typedef __attribute__((ext_vector_type(16))) _Float16 v16h;
typedef __attribute__((ext_vector_type(8)))  float    v8f;
typedef __attribute__((ext_vector_type(8)))  int      v8i_t;

#define NUM_OPS 7

// fp32 value of sigmoid(-20): the exact per-neighbor gate weight the reference
// produces in float32 (sigmoid(20) rounds to exactly 1.0f in fp32).
#define GATE_EPS 2.0611536e-9f

// ---- dynamic row-select out of the 8-VGPR f32 accumulator -----------------
__device__ __forceinline__ float sel8(v8f d, int r) {
    float v = d[0];
    v = (r == 1) ? d[1] : v;
    v = (r == 2) ? d[2] : v;
    v = (r == 3) ? d[3] : v;
    v = (r == 4) ? d[4] : v;
    v = (r == 5) ? d[5] : v;
    v = (r == 6) ? d[6] : v;
    v = (r == 7) ? d[7] : v;
    return v;
}

// C/D layout: lanes 0-15 hold rows M=0..7 (VGPR v -> M=v) of column N=lane,
// lanes 16-31 hold rows M=8..15 of column N=lane-16.  Combine with shfl_xor(16).
__device__ __forceinline__ float extract_row(v8f d, int row, bool hiHalf) {
    float v0 = sel8(d, row & 7);
    float v1 = __shfl_xor(v0, 16, 32);
    bool lowRow = row < 8;
    return (lowRow != hiHalf) ? v0 : v1;
}

__global__ __launch_bounds__(256) void C4MoEVM_kernel(
    const float* __restrict__ a_in, const float* __restrict__ b_in,
    const int* __restrict__ op_in, const float* __restrict__ recip_val,
    float* __restrict__ out, int B)
{
    __shared__ float lds_recip[256];

    const unsigned tid  = threadIdx.x;
    const unsigned lane = tid & 31u;
    const unsigned Mrow = lane & 15u;
    const bool hiHalf   = lane >= 16u;

    // ---- stage 256-entry reciprocal table via gfx1250 async global->LDS ----
    {
        unsigned lds_addr = (unsigned)(size_t)&lds_recip[tid]; // low 32b of flat = LDS offset
        unsigned voff     = tid * 4u;
        asm volatile("global_load_async_to_lds_b32 %0, %1, %2"
                     :: "v"(lds_addr), "v"(voff), "s"(recip_val) : "memory");
#if __has_builtin(__builtin_amdgcn_s_wait_asynccnt)
        __builtin_amdgcn_s_wait_asynccnt(0);
#else
        asm volatile("s_wait_asynccnt 0" ::: "memory");
#endif
    }
    __syncthreads();

    // ---- A operand: AND table as f16 16x32 (K=16..31 zero-padded) ---------
    // A k order used here (contiguous, same as B): lane<16: half q -> K=q;
    //                  lane>=16: half q -> K=q+16.
    v16h Aand;
#pragma unroll
    for (int q = 0; q < 16; ++q) {
        int k = hiHalf ? q + 16 : q;
        Aand[q] = (_Float16)((k < 16) ? (float)(Mrow & (unsigned)k) : 0.0f);
    }

    const int i     = (int)(blockIdx.x * 256u + tid);
    const int iload = (i < B) ? i : (B - 1);

    const float a   = a_in[iload];
    const float b   = b_in[iload];
    const int   opc = op_in[iload];
    const int   ia  = ((int)a) & 15;   // exact small integers
    const int   ib  = ((int)b) & 15;

    // ---- AND expert: einsum 'ij,bj->ib' as WMMA (one-hot f16 B operand) ---
    // B 32x16 f16: column n = lane&15; lanes 0-15 hold K=0..15 (half q -> K=q),
    // lanes 16-31 hold K=16..31 (all zero here since ib<16).
    float andv = 0.0f;
#pragma unroll
    for (int g = 0; g < 2; ++g) {
        const int src = (int)(lane & 15u) + 16 * g;   // element owning column n
        const int ibs = __shfl(ib, src, 32);
        const int ias = __shfl(ia, src, 32);

        const int pat    = 0x3C00 << ((ibs & 1) * 16);     // f16 1.0 in lo/hi half
        const int hi_idx = hiHalf ? 64 : (ibs >> 1);       // 64: never matches p
        v8i_t Bi;
#pragma unroll
        for (int p = 0; p < 8; ++p) Bi[p] = (hi_idx == p) ? pat : 0;
        v16h Bm = __builtin_bit_cast(v16h, Bi);

        v8f z = {0.f, 0.f, 0.f, 0.f, 0.f, 0.f, 0.f, 0.f};
        v8f Dand = __builtin_amdgcn_wmma_f32_16x16x32_f16(false, Aand, false, Bm, (short)0, z, false, false);

        const float t = extract_row(Dand, ias, hiHalf);
        andv = (((int)(lane >> 4)) == g) ? t : andv;
    }
    // exact integer identities (values <= 30, exact in fp32)
    const float orv  = (a + b) - andv;          // i|j = i+j - (i&j)
    const float xorv = (a + b) - 2.0f * andv;   // i^j = i+j - 2(i&j)

    // ---- reciprocal expert: frexp-normalize, LDS seed, 2 Newton steps -----
    int ex; float mant = frexpf(b, &ex);            // b = mant * 2^ex, mant in [0.5,1)
    int ridx = (int)rintf((mant - 0.5f) * 256.0f);
    ridx = ridx < 0 ? 0 : (ridx > 255 ? 255 : ridx);
    float y = lds_recip[ridx];
    y = y * (2.0f - mant * y);
    y = y * (2.0f - mant * y);
    const float recipv = ldexpf(y, -ex);

    // ---- routing: fp32-exact closed form of the chained SiLU eq-gates -----
    // gate(d)=1 at d==0, +eps at |d|==1, -eps at |d|==2, <1e-17 beyond.
    const float experts[NUM_OPS] = { a + b, a - b, rintf(a * b), andv, orv, xorv, recipv };
    float acc = 0.0f;
#pragma unroll
    for (int k = 0; k < NUM_OPS; ++k) {
        int ad = opc - k; ad = (ad < 0) ? -ad : ad;
        float c = (ad == 0) ? 1.0f
                : (ad == 1) ? GATE_EPS
                : (ad == 2) ? -GATE_EPS
                : 0.0f;
        acc = fmaf(c, experts[k], acc);
    }

    if (i < B) out[i] = acc;
}

extern "C" void kernel_launch(void* const* d_in, const int* in_sizes, int n_in,
                              void* d_out, int out_size, void* d_ws, size_t ws_size,
                              hipStream_t stream) {
    const float* a         = (const float*)d_in[0];
    const float* b         = (const float*)d_in[1];
    const int*   opcode    = (const int*)d_in[2];
    // d_in[3..5] = and/or/xor tables; OR/XOR derived exactly from AND + a + b
    const float* recip_val = (const float*)d_in[6];
    float* out = (float*)d_out;

    const int B = in_sizes[0];
    const int blocks = (B + 255) / 256;
    C4MoEVM_kernel<<<blocks, 256, 0, stream>>>(a, b, opcode, recip_val, out, B);
}
